// HierarchicalPatternMemory_58454504898998
// MI455X (gfx1250) — matrix operand order repaired and checked by reading, not compiled, not run on hardware
//
#include <hip/hip_runtime.h>
#include <hip/hip_bf16.h>
#include <math.h>
#include <stdint.h>

// ---------------------------------------------------------------------------
// HierarchicalPatternMemory — MI455X (gfx1250)
// Dominant cost: q = cls @ Wq^T (8192x4096x4096, 274 GFLOP) -> bf16 WMMA,
// tiles staged by the Tensor Data Mover (TENSORcnt double buffering).
// ---------------------------------------------------------------------------

typedef __attribute__((ext_vector_type(4)))  float        v4f;
typedef __attribute__((ext_vector_type(8)))  float        v8f;
typedef __attribute__((ext_vector_type(4)))  __bf16       v4bf;
typedef __attribute__((ext_vector_type(8)))  __bf16       v8bf;
typedef __attribute__((ext_vector_type(16))) __bf16       v16bf;
typedef __attribute__((ext_vector_type(4)))  unsigned int u32x4;
typedef __attribute__((ext_vector_type(4)))  int          i32x4;
typedef __attribute__((ext_vector_type(8)))  int          i32x8;

constexpr int Bn = 8192;
constexpr int Dn = 4096;
constexpr int Kn = 8;
constexpr int Mn = 4;
constexpr int Pn = 256;
constexpr int NPROTO = Kn + Kn * Mn;   // 40

// ---- TDM availability / arity probing (per CDNA5_HIP.md) ------------------
#if defined(__AMDGCN__) && __has_builtin(__builtin_amdgcn_tensor_load_to_lds) && \
    __has_builtin(__builtin_amdgcn_s_wait_tensorcnt)
#define HAVE_TDM 1
#else
#define HAVE_TDM 0
#endif
#if HAVE_TDM && __has_include(<hip/amd_detail/amd_gfx1250_TDM.h>)
#define TDM_6ARG 1
#else
#define TDM_6ARG 0
#endif

// ------------------------------ reductions ---------------------------------

__device__ __forceinline__ float waveReduceSum(float v) {
#pragma unroll
  for (int off = 16; off > 0; off >>= 1) v += __shfl_xor(v, off, 32);
  return v;
}

__device__ __forceinline__ void blockReduce2(float& a, float& b, float* red, int tid) {
  a = waveReduceSum(a);
  b = waveReduceSum(b);
  const int w = tid >> 5;
  if ((tid & 31) == 0) { red[w] = a; red[8 + w] = b; }
  __syncthreads();
  float sa = 0.f, sb = 0.f;
#pragma unroll
  for (int i = 0; i < 8; ++i) { sa += red[i]; sb += red[8 + i]; }
  a = sa; b = sb;
  __syncthreads();
}

// -------------------- kernel: f32 -> bf16 pre-conversion -------------------
// 8 elements per thread; enables pure-byte TDM tile DMA in the GEMM.

__global__ __launch_bounds__(256) void k_cvt_bf16(
    const float* __restrict__ src, __bf16* __restrict__ dst)
{
  const size_t i = ((size_t)blockIdx.x * 256 + threadIdx.x) * 8;
  v4f a = *(const v4f*)&src[i];
  v4f b = *(const v4f*)&src[i + 4];
  v8bf o;
#pragma unroll
  for (int j = 0; j < 4; ++j) { o[j] = (__bf16)a[j]; o[4 + j] = (__bf16)b[j]; }
  *(v8bf*)&dst[i] = o;
}

// -------------------- kernel 0a: l2-normalize prototypes -------------------

__global__ __launch_bounds__(256) void k_norm_protos(
    const float* __restrict__ coarse, const float* __restrict__ fine,
    float* __restrict__ protosN)
{
  __shared__ float red[16];
  const int p = blockIdx.x, tid = threadIdx.x;
  const float* src = (p < Kn) ? (coarse + (size_t)p * Dn)
                              : (fine + (size_t)(p - Kn) * Dn);
  float ss = 0.f, dummy = 0.f;
  for (int d = tid; d < Dn; d += 256) { float x = src[d]; ss += x * x; }
  blockReduce2(ss, dummy, red, tid);
  const float inv = 1.0f / fmaxf(sqrtf(ss), 1e-12f);
  float* dst = protosN + (size_t)p * Dn;
  for (int d = tid; d < Dn; d += 256) dst[d] = src[d] * inv;
}

// ---------------- kernel 0b: fine_prompts = fine @ Wp^T + bp ---------------

__global__ __launch_bounds__(256) void k_fine_prompts(
    const float* __restrict__ fine, const float* __restrict__ Wp,
    const float* __restrict__ bp, float* __restrict__ fpOut)
{
  __shared__ float rowv[Dn];
  const int r = blockIdx.x, tid = threadIdx.x;
  const float* src = fine + (size_t)r * Dn;
  for (int d = tid; d < Dn; d += 256) rowv[d] = src[d];
  __syncthreads();
  const float* w = Wp + (size_t)tid * Dn;
  float acc = 0.f;
  for (int d = 0; d < Dn; ++d) acc += rowv[d] * w[d];
  fpOut[r * Pn + tid] = acc + bp[tid];
}

// -------------- kernel 1: q = cls @ Wq^T + bq  (bf16 WMMA + TDM) -----------

constexpr int BM = 128, BN = 128, BK = 32, LDT = 40;  // LDT: padded halves/row
constexpr int TILEH = BM * LDT;                       // halves per LDS tile

#if HAVE_TDM
// Issue one 2D TDM tile DMA: rows x BK bf16 elements, row stride = Dn elems,
// LDS rows padded from 32 halves (16 DW) to 40 halves (+4 DW) => LDT layout.
// D# packing per CDNA5 ISA §8.3/§8.4.
__device__ __forceinline__ void tdm_load_tile(unsigned lds_off,
                                              const __bf16* gptr) {
  const unsigned long long ga = (unsigned long long)(uintptr_t)gptr;
  u32x4 g0;
  g0[0] = 1u;                                        // count=1, gather off
  g0[1] = lds_off;                                   // lds_addr (bytes)
  g0[2] = (unsigned)(ga & 0xFFFFFFFFu);              // global_addr[31:0]
  g0[3] = (unsigned)((ga >> 32) & 0x01FFFFFFu)       // global_addr[56:32]
          | (2u << 30);                              // type=2 ("image")
  i32x8 g1;
  g1[0] = (int)((1u << 16)       // data_size = 2 bytes
              | (1u << 20)       // pad_enable
              | (3u << 22)       // pad_interval: every 16 DWORDs
              | (3u << 25));     // pad_amount: 4 DWORDs -> 40-half rows
  g1[1] = (int)(0u);                      // barrier addr 0 | tensor_dim0 lo=0
  g1[2] = (int)(0x4000u);                 // tensor_dim0 hi (2^30) | dim1 lo=0
  g1[3] = (int)(0x4000u | ((unsigned)BK << 16)); // dim1 hi (2^30) | tile_dim0
  g1[4] = (int)BM;                        // tile_dim1 | tile_dim2=0
  g1[5] = (int)Dn;                        // tensor_dim0_stride lo
  g1[6] = 0;                              // stride hi | dim1_stride lo
  g1[7] = 0;                              // dim1_stride hi
  i32x4 z4 = {0, 0, 0, 0};
#if TDM_6ARG
  i32x8 z8 = {0, 0, 0, 0, 0, 0, 0, 0};
  __builtin_amdgcn_tensor_load_to_lds(g0, g1, z4, z4, z8, 0);
#else
  __builtin_amdgcn_tensor_load_to_lds(g0, g1, z4, z4, 0);
#endif
}
#endif

__global__ __launch_bounds__(256) void k_gemm_q(
    const __bf16* __restrict__ A,   // (B, D) bf16
    const __bf16* __restrict__ W,   // (D, D) bf16
    const float* __restrict__ bq,
    float* __restrict__ Q)          // (B, D) f32
{
#if HAVE_TDM
  __shared__ __bf16 smem[4 * TILEH];   // [buf0:A,W][buf1:A,W]
#else
  __shared__ __bf16 smem[2 * TILEH];
#endif

  const int tid  = threadIdx.x;
  const int lane = tid & 31;
  const int wave = tid >> 5;
  const int wm = wave & 1;           // 2 waves along M
  const int wn = wave >> 1;          // 4 waves along N
  const int m0 = blockIdx.y * BM;
  const int n0 = blockIdx.x * BN;
  const int hl = lane >> 4;
  const int l15 = lane & 15;

  v8f acc[4][2] = {};
  const int nk = Dn / BK;

#if HAVE_TDM
  if (wave == 0) {
    tdm_load_tile((unsigned)(uintptr_t)&smem[0 * TILEH], &A[(size_t)m0 * Dn]);
    tdm_load_tile((unsigned)(uintptr_t)&smem[1 * TILEH], &W[(size_t)n0 * Dn]);
  }
#endif

  for (int it = 0; it < nk; ++it) {
    const int k0 = it * BK;
#if HAVE_TDM
    const int cur = it & 1;
    const __bf16* tA = &smem[(cur * 2 + 0) * TILEH];
    const __bf16* tW = &smem[(cur * 2 + 1) * TILEH];
    if (wave == 0) __builtin_amdgcn_s_wait_tensorcnt(0);  // buffers cur ready
    __syncthreads();   // publish LDS to all waves; also: compute(it-1) done,
                       // so buffer cur^1 may be overwritten by the next DMA
    if (wave == 0 && it + 1 < nk) {
      const int nxt = cur ^ 1;
      tdm_load_tile((unsigned)(uintptr_t)&smem[(nxt * 2 + 0) * TILEH],
                    &A[(size_t)m0 * Dn + k0 + BK]);
      tdm_load_tile((unsigned)(uintptr_t)&smem[(nxt * 2 + 1) * TILEH],
                    &W[(size_t)n0 * Dn + k0 + BK]);
    }
#else
    const __bf16* tA = &smem[0];
    const __bf16* tW = &smem[TILEH];
    __syncthreads();
    {
      __bf16* sA = &smem[0];
      __bf16* sW = &smem[TILEH];
#pragma unroll
      for (int p = 0; p < 2; ++p) {
        const int r = (tid >> 2) + p * 64;     // 0..127
        const int c = (tid & 3) * 8;           // halves
        *(v8bf*)&sA[r * LDT + c] = *(const v8bf*)&A[(size_t)(m0 + r) * Dn + k0 + c];
        *(v8bf*)&sW[r * LDT + c] = *(const v8bf*)&W[(size_t)(n0 + r) * Dn + k0 + c];
      }
    }
    __syncthreads();
    if (k0 + BK < Dn) {
      __builtin_prefetch(&A[(size_t)(m0 + (tid >> 2)) * Dn + k0 + BK], 0, 1);
      __builtin_prefetch(&W[(size_t)(n0 + (tid >> 2)) * Dn + k0 + BK], 0, 1);
    }
#endif

    // A fragment (16x32 bf16): lane m holds K[8h..8h+7], K[16+8h..16+8h+7]
    v16bf afr[4];
#pragma unroll
    for (int mt = 0; mt < 4; ++mt) {
      const __bf16* base = &tA[(wm * 64 + mt * 16 + l15) * LDT];
      v8bf lo = *(const v8bf*)(base + hl * 8);
      v8bf hi = *(const v8bf*)(base + 16 + hl * 8);
#pragma unroll
      for (int i = 0; i < 8; ++i) { afr[mt][i] = lo[i]; afr[mt][8 + i] = hi[i]; }
    }
    // B fragment (32x16 bf16): lane n holds K[8h..8h+7], K[16+8h..16+8h+7] of column n
    v16bf bfr[2];
#pragma unroll
    for (int nt = 0; nt < 2; ++nt) {
      const __bf16* base = &tW[(wn * 32 + nt * 16 + l15) * LDT];
      v8bf lo = *(const v8bf*)(base + hl * 8);
      v8bf hi = *(const v8bf*)(base + 16 + hl * 8);
#pragma unroll
      for (int i = 0; i < 8; ++i) { bfr[nt][i] = lo[i]; bfr[nt][8 + i] = hi[i]; }
    }

#pragma unroll
    for (int mt = 0; mt < 4; ++mt)
#pragma unroll
      for (int nt = 0; nt < 2; ++nt)
        acc[mt][nt] = __builtin_amdgcn_wmma_f32_16x16x32_bf16(
            false, afr[mt], false, bfr[nt], (short)0, acc[mt][nt], false, false);
  }

  // epilogue: VGPR r -> M=r (lanes 0-15) / M=r+8 (lanes 16-31), N=l15
#pragma unroll
  for (int mt = 0; mt < 4; ++mt)
#pragma unroll
    for (int nt = 0; nt < 2; ++nt) {
      const int col = n0 + wn * 32 + nt * 16 + l15;
      const float bias = bq[col];
#pragma unroll
      for (int r = 0; r < 8; ++r) {
        const int row = m0 + wm * 64 + mt * 16 + r + hl * 8;
        Q[(size_t)row * Dn + col] = acc[mt][nt][r] + bias;
      }
    }
}

// ------- kernel 2: LayerNorm + l2norm + sims + softmax + top-p per row -----

__global__ __launch_bounds__(256) void k_row(
    const float* __restrict__ Q,
    const float* __restrict__ gamma,
    const float* __restrict__ beta,
    const float* __restrict__ protosN,
    float* __restrict__ coarseOut,
    float* __restrict__ fineOut,
    float* __restrict__ combOut)
{
  __shared__ float ln[Dn];
  __shared__ float red[16];
  __shared__ float sims[NPROTO];

  const int row = blockIdx.x;
  const int tid = threadIdx.x;
  const float* qrow = Q + (size_t)row * Dn;

  float qv[16];
  float s0 = 0.f, s1 = 0.f;
#pragma unroll
  for (int i = 0; i < 16; ++i) {
    float x = qrow[tid + 256 * i];
    qv[i] = x; s0 += x; s1 += x * x;
  }
  blockReduce2(s0, s1, red, tid);
  const float mu = s0 * (1.0f / Dn);
  const float var = s1 * (1.0f / Dn) - mu * mu;
  const float rstd = rsqrtf(var + 1e-5f);

  float ss = 0.f, dummy = 0.f;
#pragma unroll
  for (int i = 0; i < 16; ++i) {
    const int d = tid + 256 * i;
    const float v = gamma[d] * (qv[i] - mu) * rstd + beta[d];
    ln[d] = v; ss += v * v;
  }
  blockReduce2(ss, dummy, red, tid);
  const float inv = 1.0f / fmaxf(sqrtf(ss), 1e-12f);

  const int lane = tid & 31, wave = tid >> 5;
  for (int p = wave; p < NPROTO; p += 8) {
    const float* pr = protosN + (size_t)p * Dn;
    float acc = 0.f;
    for (int i = lane; i < Dn; i += 32) acc += ln[i] * pr[i];
    acc = waveReduceSum(acc);
    if (lane == 0) sims[p] = acc * inv;
  }
  __syncthreads();

  if (tid == 0) {
    float cw[Kn];
    float mx = -1e30f;
#pragma unroll
    for (int k = 0; k < Kn; ++k) { cw[k] = sims[k] * 2.0f; mx = fmaxf(mx, cw[k]); }
    float sum = 0.f;
#pragma unroll
    for (int k = 0; k < Kn; ++k) { cw[k] = expf(cw[k] - mx); sum += cw[k]; }
#pragma unroll
    for (int k = 0; k < Kn; ++k) cw[k] /= sum;

    int order[Kn];
#pragma unroll
    for (int k = 0; k < Kn; ++k) order[k] = k;
    for (int a = 1; a < Kn; ++a) {
      const int o = order[a]; const float v = cw[o];
      int b = a - 1;
      while (b >= 0 && cw[order[b]] < v) { order[b + 1] = order[b]; --b; }
      order[b + 1] = o;
    }
    float csum = 0.f; int num_sel = 0;
#pragma unroll
    for (int a = 0; a < Kn; ++a) { csum += cw[order[a]]; if (csum <= 0.9f) ++num_sel; }
    bool sel[Kn];
#pragma unroll
    for (int a = 0; a < Kn; ++a) sel[order[a]] = (a <= num_sel);

    float* co = coarseOut + (size_t)row * Kn;
    float* fo = fineOut + (size_t)row * (Kn * Mn);
    float* cb = combOut + (size_t)row * (Kn * Mn);
#pragma unroll
    for (int k = 0; k < Kn; ++k) {
      co[k] = cw[k];
      float f[Mn]; float fm = -1e30f;
#pragma unroll
      for (int m = 0; m < Mn; ++m) { f[m] = sims[Kn + k * Mn + m] * 2.0f; fm = fmaxf(fm, f[m]); }
      float fs = 0.f;
#pragma unroll
      for (int m = 0; m < Mn; ++m) { f[m] = expf(f[m] - fm); fs += f[m]; }
      const float selk = sel[k] ? 1.0f : 0.0f;
#pragma unroll
      for (int m = 0; m < Mn; ++m) {
        const float fw = (f[m] / fs) * selk;
        fo[k * Mn + m] = fw;
        cb[k * Mn + m] = cw[k] * fw;
      }
    }
  }
}

// ---------- kernel 3: prompt[b,p] = sum_km combined * fine_prompts ---------

__global__ __launch_bounds__(256) void k_prompt(
    const float* __restrict__ comb, const float* __restrict__ fp,
    float* __restrict__ prompt)
{
  __shared__ float c[Kn * Mn];
  const int row = blockIdx.x;
  const int tid = threadIdx.x;
  if (tid < Kn * Mn) c[tid] = comb[(size_t)row * (Kn * Mn) + tid];
  __syncthreads();
  float acc = 0.f;
#pragma unroll
  for (int t = 0; t < Kn * Mn; ++t) acc += c[t] * fp[t * Pn + tid];
  prompt[(size_t)row * Pn + tid] = acc;
}

// ------------------------------- launcher ----------------------------------

extern "C" void kernel_launch(void* const* d_in, const int* in_sizes, int n_in,
                              void* d_out, int out_size, void* d_ws, size_t ws_size,
                              hipStream_t stream) {
  const float* cls    = (const float*)d_in[0];
  const float* Wq     = (const float*)d_in[1];
  const float* bq     = (const float*)d_in[2];
  const float* gamma  = (const float*)d_in[3];
  const float* beta   = (const float*)d_in[4];
  const float* cproto = (const float*)d_in[5];
  const float* fproto = (const float*)d_in[6];
  const float* Wp     = (const float*)d_in[7];
  const float* bp     = (const float*)d_in[8];

  float* out    = (float*)d_out;                 // [prompt | coarse_w | fine_w]
  float* prompt = out;
  float* coarse = out + (size_t)Bn * Pn;
  float* fine   = coarse + (size_t)Bn * Kn;

  float* ws      = (float*)d_ws;
  float* q       = ws;                                      // B*D f32
  float* protosN = q + (size_t)Bn * Dn;                     // 40*D f32
  float* fp      = protosN + (size_t)NPROTO * Dn;           // 32*P f32
  float* comb    = fp + (size_t)(Kn * Mn) * Pn;             // B*32 f32
  __bf16* Abf    = (__bf16*)(comb + (size_t)Bn * Kn * Mn);  // B*D bf16
  __bf16* Wbf    = Abf + (size_t)Bn * Dn;                   // D*D bf16
  (void)in_sizes; (void)n_in; (void)out_size; (void)ws_size;

  // one-shot bf16 pre-conversion so the GEMM tiles are pure-byte TDM copies
  k_cvt_bf16<<<(Bn * (size_t)Dn) / 2048, 256, 0, stream>>>(cls, Abf);
  k_cvt_bf16<<<(Dn * (size_t)Dn) / 2048, 256, 0, stream>>>(Wq, Wbf);

  k_norm_protos<<<NPROTO, 256, 0, stream>>>(cproto, fproto, protosN);
  k_fine_prompts<<<Kn * Mn, 256, 0, stream>>>(fproto, Wp, bp, fp);

  dim3 g1(Dn / BN, Bn / BM);   // (32, 64)
  k_gemm_q<<<g1, 256, 0, stream>>>(Abf, Wbf, bq, q);

  k_row<<<Bn, 256, 0, stream>>>(q, gamma, beta, protosN, coarse, fine, comb);
  k_prompt<<<Bn, 256, 0, stream>>>(comb, fp, prompt);
}
